// CustomAttention_84593675862739
// MI455X (gfx1250) — matrix operand order repaired and checked by reading, not compiled, not run on hardware
//
#include <hip/hip_runtime.h>
#include <hip/hip_bf16.h>

#define DD 768
#define LL 512
#define BB 64
#define MT 64              // rows of hidden per workgroup in the WMMA kernel
#define LDS_STRIDE 776     // 768 + 8 bf16 pad to avoid LDS bank conflicts

typedef __attribute__((ext_vector_type(16))) __bf16 bf16x16;
typedef __attribute__((ext_vector_type(4)))  __bf16 bf16x4;
typedef __attribute__((ext_vector_type(8)))  float  floatx8;

union AFrag { bf16x16 v; uint4 u[2]; };

// Native CDNA5 V_TANH_F32 if the toolchain exposes it; libm fallback otherwise.
__device__ __forceinline__ float fast_tanh(float x) {
#if defined(__has_builtin)
#if __has_builtin(__builtin_amdgcn_tanhf)
    return __builtin_amdgcn_tanhf(x);
#else
    return tanhf(x);
#endif
#else
    return tanhf(x);
#endif
}

// ---------------------------------------------------------------------------
// Kernel 0: convert W [768x768] fp32 -> bf16 into workspace (stays L2-resident)
// ---------------------------------------------------------------------------
__global__ __launch_bounds__(256) void ca_cvt_w_kernel(const float* __restrict__ W,
                                                       __bf16* __restrict__ Wb) {
    int i = (blockIdx.x * 256 + threadIdx.x) * 4;   // 589824 elems / 4 / 256 = 576 blocks
    float4 f = *(const float4*)(W + i);
    bf16x4 o;
    o[0] = (__bf16)f.x; o[1] = (__bf16)f.y; o[2] = (__bf16)f.z; o[3] = (__bf16)f.w;
    *(bf16x4*)(Wb + i) = o;
}

// ---------------------------------------------------------------------------
// Kernel 1: scores[m] = sum_e tanh( h[m,:]·W[e,:] + b[e] ) * c[e]
// Fused GEMM (bf16 WMMA) + epilogue reduction. proj is never materialized.
// Block = 256 threads (8 waves). Each workgroup: 64 rows, full K=768 staged
// in LDS as bf16. Waves 0-3 handle N-tiles 0..23, waves 4-7 handle 24..47.
// ---------------------------------------------------------------------------
__global__ __launch_bounds__(256) void ca_scores_kernel(const float* __restrict__ hidden,
                                                        const float* __restrict__ bias,
                                                        const float* __restrict__ cvec,
                                                        const __bf16* __restrict__ Wb,
                                                        float* __restrict__ scores) {
    extern __shared__ __bf16 lds[];                       // MT * LDS_STRIDE bf16
    float* partial = (float*)(lds + (size_t)MT * LDS_STRIDE);  // [MT][2] floats

    const int t  = threadIdx.x;
    const int m0 = blockIdx.x * MT;

    // ---- Phase 1: load 64x768 fp32 rows of hidden, convert to bf16 in LDS ----
    // 49152 floats -> 12288 float4 -> 48 per thread
    #pragma unroll 4
    for (int i = 0; i < (MT * DD) / (256 * 4); ++i) {
        int e4 = i * 256 + t;
        int e  = e4 * 4;
        int r  = e / DD;
        int k  = e % DD;
        float4 f = *(const float4*)(hidden + (size_t)m0 * DD + e);
        bf16x4 o;
        o[0] = (__bf16)f.x; o[1] = (__bf16)f.y; o[2] = (__bf16)f.z; o[3] = (__bf16)f.w;
        *(bf16x4*)(lds + r * LDS_STRIDE + k) = o;
    }
    __syncthreads();

    // ---- Phase 2: WMMA ----
    const int wave  = t >> 5;
    const int lane  = t & 31;
    const int laneL = lane & 15;
    const int hi    = lane >> 4;        // lane half (0: lanes 0-15, 1: lanes 16-31)
    const int wrow  = wave & 3;         // which 16-row group this wave owns
    const int ngrp  = wave >> 2;        // N-tile split: 0 -> tiles 0..23, 1 -> 24..47
    const int rowBase = wrow * 16;

    float sacc[8];
    #pragma unroll
    for (int v = 0; v < 8; ++v) sacc[v] = 0.0f;

    // A-fragment LDS base (element units): row = rowBase + laneL
    // lanes 0-15 read K {k0..k0+7, k0+16..k0+23}; lanes 16-31 read {k0+8.., k0+24..}
    const int aRow = (rowBase + laneL) * LDS_STRIDE + hi * 8;

    const int ntBeg = ngrp * 24;
    for (int nt = ntBeg; nt < ntBeg + 24; ++nt) {
        const int n0 = nt * 16;
        const int e  = n0 + laneL;                    // output column this lane owns
        const float be = bias[e];
        const float ce = cvec[e];

        floatx8 acc = {0.f, 0.f, 0.f, 0.f, 0.f, 0.f, 0.f, 0.f};

        // B-fragment: lane holds column e of W^T = W[e, k] contiguous in k.
        const __bf16* bp = Wb + (size_t)e * DD + hi * 8;

        #pragma unroll 4
        for (int k0 = 0; k0 < DD; k0 += 32) {
            AFrag af, bf;
            const __bf16* ap = lds + aRow + k0;
            af.u[0] = *(const uint4*)(ap);
            af.u[1] = *(const uint4*)(ap + 16);
            bf.u[0] = *(const uint4*)(bp + k0);
            bf.u[1] = *(const uint4*)(bp + k0 + 16);
            acc = __builtin_amdgcn_wmma_f32_16x16x32_bf16(
                      false, af.v, false, bf.v, (short)0, acc, false, false);
        }

        // Epilogue: tanh + dot with context vector, accumulated per row.
        // C layout: VGPR v, lane -> element (M = v + 8*hi, N = laneL)
        #pragma unroll
        for (int v = 0; v < 8; ++v) {
            float p = fast_tanh(acc[v] + be);
            sacc[v] += p * ce;
        }
    }

    // ---- Reduce over the 16 lanes of each half (columns of the tile) ----
    #pragma unroll
    for (int v = 0; v < 8; ++v) {
        float s = sacc[v];
        s += __shfl_xor(s, 1);
        s += __shfl_xor(s, 2);
        s += __shfl_xor(s, 4);
        s += __shfl_xor(s, 8);
        if (laneL == 0) partial[(rowBase + hi * 8 + v) * 2 + ngrp] = s;
    }
    __syncthreads();

    // ---- Combine the two N-half partials, write raw scores ----
    if (t < MT) scores[m0 + t] = partial[t * 2 + 0] + partial[t * 2 + 1];
}

// ---------------------------------------------------------------------------
// Kernel 2: per-batch softmax over L=512 + weighted sum over hidden.
// grid (64, 3): blockIdx.x = batch, blockIdx.y = 256-wide D chunk.
// Softmax is recomputed from raw scores in every block (no cross-block race);
// only the y==0 block writes attention weights to d_out.
// ---------------------------------------------------------------------------
__global__ __launch_bounds__(256) void ca_softmax_wsum_kernel(const float* __restrict__ hidden,
                                                              const float* __restrict__ scores,
                                                              float* __restrict__ out) {
    __shared__ float wbuf[LL];
    __shared__ float rbuf[8];

    const int b    = blockIdx.x;
    const int t    = threadIdx.x;
    const int lane = t & 31;
    const int wv   = t >> 5;

    const float* sc = scores + (size_t)b * LL;
    float s0 = sc[t];
    float s1 = sc[t + 256];

    // max over 512
    float m = fmaxf(s0, s1);
    #pragma unroll
    for (int msk = 1; msk < 32; msk <<= 1) m = fmaxf(m, __shfl_xor(m, msk));
    if (lane == 0) rbuf[wv] = m;
    __syncthreads();
    float M = rbuf[0];
    #pragma unroll
    for (int i = 1; i < 8; ++i) M = fmaxf(M, rbuf[i]);
    __syncthreads();

    // exp + sum (native v_exp_f32 path)
    float e0 = __expf(s0 - M);
    float e1 = __expf(s1 - M);
    float s = e0 + e1;
    #pragma unroll
    for (int msk = 1; msk < 32; msk <<= 1) s += __shfl_xor(s, msk);
    if (lane == 0) rbuf[wv] = s;
    __syncthreads();
    float S = 0.0f;
    #pragma unroll
    for (int i = 0; i < 8; ++i) S += rbuf[i];
    const float inv = 1.0f / S;

    const float w0 = e0 * inv;
    const float w1 = e1 * inv;
    wbuf[t]       = w0;
    wbuf[t + 256] = w1;
    if (blockIdx.y == 0) {
        float* aw = out + (size_t)BB * DD + (size_t)b * LL;
        aw[t]       = w0;
        aw[t + 256] = w1;
    }
    __syncthreads();

    // weighted sum over L for this block's 256 columns
    const int d = blockIdx.y * 256 + t;
    const float* hb = hidden + (size_t)b * LL * DD + d;
    float acc = 0.0f;
    #pragma unroll 4
    for (int l = 0; l < LL; ++l) acc += wbuf[l] * hb[(size_t)l * DD];
    out[(size_t)b * DD + d] = acc;
}

// ---------------------------------------------------------------------------
// Launch. Inputs: [0]=hidden_states(B,L,D) [1]=W(D,D) [2]=b(D) [3]=context(D,1)
// Output: weighted_output(B,D) then attention_weights(B,L), flat fp32.
// Workspace: W as bf16 (1.18 MB) then raw scores (128 KB).
// ---------------------------------------------------------------------------
extern "C" void kernel_launch(void* const* d_in, const int* in_sizes, int n_in,
                              void* d_out, int out_size, void* d_ws, size_t ws_size,
                              hipStream_t stream) {
    const float* hidden = (const float*)d_in[0];
    const float* W      = (const float*)d_in[1];
    const float* bias   = (const float*)d_in[2];
    const float* cvec   = (const float*)d_in[3];
    float* out          = (float*)d_out;

    __bf16* Wb    = (__bf16*)d_ws;
    float* scores = (float*)((char*)d_ws + (size_t)DD * DD * sizeof(__bf16));

    // 768*768/4/256 = 576 blocks
    ca_cvt_w_kernel<<<576, 256, 0, stream>>>(W, Wb);

    const int mBlocks = (BB * LL) / MT;                        // 512
    const size_t smem = (size_t)MT * LDS_STRIDE * sizeof(__bf16) + MT * 2 * sizeof(float);
    ca_scores_kernel<<<mBlocks, 256, smem, stream>>>(hidden, bias, cvec, Wb, scores);

    ca_softmax_wsum_kernel<<<dim3(BB, 3), 256, 0, stream>>>(hidden, scores, out);
}
